// PhiAttention_6356551598806
// MI455X (gfx1250) — matrix operand order repaired and checked by reading, not compiled, not run on hardware
//
#include <hip/hip_runtime.h>
#include <hip/hip_bf16.h>

// ---------------------------------------------------------------------------
// PhiAttention for MI455X (gfx1250): all matmuls on v_wmma_f32_16x16x32_f16,
// GEMM B-tiles staged in LDS via async global->LDS DMA (ASYNCcnt pipeline).
// B=2, S=2048, E=HID=1024, H=16, D=64.
// ---------------------------------------------------------------------------

typedef _Float16 v16h __attribute__((ext_vector_type(16)));
typedef _Float16 h8   __attribute__((ext_vector_type(8)));
typedef float    v8f  __attribute__((ext_vector_type(8)));

// pointee type the async-LDS builtin expects (per hipcc diagnostic):
typedef int v4i_vs __attribute__((vector_size(16)));
// correctly-spelled LDS address space (attribute BEFORE declarator name)
typedef __attribute__((address_space(3))) void as3_void;

union AF { v16h v; h8 h[2]; };

static __device__ __forceinline__ h8 ld8g(const _Float16* p) { return *(const h8*)p; }

// ds_swizzle group-of-32 xor patterns: offset = (xor<<10) | (or<<5) | and
#define SWZ_XOR1(x) __int_as_float(__builtin_amdgcn_ds_swizzle(__float_as_int(x), 0x041F))
#define SWZ_XOR2(x) __int_as_float(__builtin_amdgcn_ds_swizzle(__float_as_int(x), 0x081F))
#define SWZ_XOR4(x) __int_as_float(__builtin_amdgcn_ds_swizzle(__float_as_int(x), 0x101F))
#define SWZ_XOR8(x) __int_as_float(__builtin_amdgcn_ds_swizzle(__float_as_int(x), 0x201F))

// --- CDNA5 async global->LDS copy (ASYNCcnt) -------------------------------
static __device__ __forceinline__ void async_ld16(const _Float16* g, _Float16* l) {
#if __has_builtin(__builtin_amdgcn_global_load_async_to_lds_b128)
    __builtin_amdgcn_global_load_async_to_lds_b128((v4i_vs*)g, (v4i_vs*)l, 0, 0);
#else
    unsigned laddr = (unsigned)(unsigned long long)(as3_void*)l;
    asm volatile("global_load_async_to_lds_b128 %0, %1, off"
                 :: "v"(laddr), "v"(g) : "memory");
#endif
}

static __device__ __forceinline__ void wait_async0() {
#if __has_builtin(__builtin_amdgcn_s_wait_asynccnt)
    __builtin_amdgcn_s_wait_asynccnt(0);
#else
    asm volatile("s_wait_asynccnt 0x0" ::: "memory");
#endif
}

// ---------------------------------------------------------------------------
// fp32 -> f16 conversion
// ---------------------------------------------------------------------------
__global__ void cvt_kernel(const float* __restrict__ s, _Float16* __restrict__ d, int n) {
    int i = blockIdx.x * blockDim.x + threadIdx.x;
    int stride = gridDim.x * blockDim.x;
    for (; i < n; i += stride) d[i] = (_Float16)s[i];
}

// ---------------------------------------------------------------------------
// Generic GEMM: C[m][n] = sum_k A[m][k] * Bw[n][k]   (M=4096, N=1024, K=1024)
// Block = 8 waves; block tile 256(M) x 64(N); wave tile 32 x 64.
// B-tile (64n x 32k) double-buffered in LDS via async global->LDS loads:
//   each of 256 threads DMA's one 16-byte chunk per K-step, overlapping the
//   WMMAs on the current buffer. One s_wait_asynccnt + one barrier per step.
// mode 0: store f16 to dst laid out [b][h][s][d]           (Q / K)
// mode 2: add bias[n], store f16 transposed [b][h][d][s]   (V)
// mode 3: add bias[n], store f32 row-major [m][n]          (output proj)
// ---------------------------------------------------------------------------
#define BSTRIDE 40   // halfs per LDS B row (80B: 16B-aligned, conflict-skewed)

__global__ __launch_bounds__(256) void gemm_wmma_kernel(
    const _Float16* __restrict__ A, const _Float16* __restrict__ Bw,
    const float* __restrict__ bias, void* __restrict__ dst, int mode)
{
    __shared__ _Float16 ldsB[2][64 * BSTRIDE];
    const int tid  = threadIdx.x;
    const int lane = tid & 31;
    const int wave = tid >> 5;
    const int l16  = lane & 15;
    const int hf   = (lane >> 4) & 1;
    const int m0   = blockIdx.y * 256 + wave * 32;
    const int n0   = blockIdx.x * 64;

    // async DMA chunk owned by this thread: row cn (0..63), 8-half chunk ck
    const int cn = tid >> 2;
    const int ck = (tid & 3) * 8;
    const _Float16* gchunk = Bw + (size_t)(n0 + cn) * 1024 + ck;
    _Float16* lchunk0 = &ldsB[0][cn * BSTRIDE + ck];
    _Float16* lchunk1 = &ldsB[1][cn * BSTRIDE + ck];

    v8f acc[8] = {};
    const _Float16* arow0 = A + (size_t)(m0 + l16) * 1024 + 8 * hf;
    const _Float16* arow1 = A + (size_t)(m0 + 16 + l16) * 1024 + 8 * hf;

    async_ld16(gchunk, lchunk0);                 // prologue: fill buf 0

    int buf = 0;
    for (int k0 = 0; k0 < 1024; k0 += 32, buf ^= 1) {
        wait_async0();                           // my chunk of buf landed
        __syncthreads();                         // all chunks landed; prev reads of buf^1 done
        if (k0 + 32 < 1024)
            async_ld16(gchunk + k0 + 32, buf ? lchunk0 : lchunk1);

        AF a0, a1;
        a0.h[0] = ld8g(arow0 + k0); a0.h[1] = ld8g(arow0 + k0 + 16);
        a1.h[0] = ld8g(arow1 + k0); a1.h[1] = ld8g(arow1 + k0 + 16);
        #pragma unroll
        for (int f = 0; f < 4; ++f) {
            const _Float16* bl = &ldsB[buf][(f * 16 + l16) * BSTRIDE + 8 * hf];
            AF bb;
            bb.h[0] = *(const h8*)bl;
            bb.h[1] = *(const h8*)(bl + 16);
            acc[f]     = __builtin_amdgcn_wmma_f32_16x16x32_f16(
                false, a0.v, false, bb.v, (short)0, acc[f], false, false);
            acc[4 + f] = __builtin_amdgcn_wmma_f32_16x16x32_f16(
                false, a1.v, false, bb.v, (short)0, acc[4 + f], false, false);
        }
    }

    #pragma unroll
    for (int g = 0; g < 2; ++g) {
        const int mb = m0 + g * 16;
        if (mode == 0) {
            _Float16* q = (_Float16*)dst;
            #pragma unroll
            for (int f = 0; f < 4; ++f) {
                int n = n0 + f * 16 + l16;
                int h = n >> 6, dd = n & 63;
                #pragma unroll
                for (int r = 0; r < 8; ++r) {
                    int m = mb + r + 8 * hf;
                    int b = m >> 11, s = m & 2047;
                    q[(((size_t)b * 16 + h) * 2048 + s) * 64 + dd] = (_Float16)acc[g * 4 + f][r];
                }
            }
        } else if (mode == 2) {
            _Float16* vt = (_Float16*)dst;
            int b  = mb >> 11;
            int sb = (mb & 2047) + 8 * hf;
            #pragma unroll
            for (int f = 0; f < 4; ++f) {
                int n = n0 + f * 16 + l16;
                float bn = bias[n];
                h8 pack;
                #pragma unroll
                for (int r = 0; r < 8; ++r) pack[r] = (_Float16)(acc[g * 4 + f][r] + bn);
                *(h8*)&vt[(((size_t)b * 16 + (n >> 6)) * 64 + (n & 63)) * 2048 + sb] = pack;
            }
        } else {
            float* out = (float*)dst;
            #pragma unroll
            for (int f = 0; f < 4; ++f) {
                int n = n0 + f * 16 + l16;
                float bn = bias[n];
                #pragma unroll
                for (int r = 0; r < 8; ++r) {
                    int m = mb + r + 8 * hf;
                    out[(size_t)m * 1024 + n] = acc[g * 4 + f][r] + bn;
                }
            }
        }
    }
}

// ---------------------------------------------------------------------------
// Interleaved RoPE on f16 tensor laid out [B*H][S][64]. 2^21 threads total:
// idx = (bh << 16) | (s << 5) | pair_i
// ---------------------------------------------------------------------------
__global__ void rope_kernel(_Float16* __restrict__ t) {
    int idx = blockIdx.x * blockDim.x + threadIdx.x;
    int i  = idx & 31;
    int s  = (idx >> 5) & 2047;
    int bh = idx >> 16;
    size_t base = ((size_t)bh * 2048 + s) * 64 + 2 * i;
    float ang = (float)s * __expf(-9.210340371976184f * (float)i / 32.0f);
    float sn, cs;
    __sincosf(ang, &sn, &cs);
    float t1 = (float)t[base], t2 = (float)t[base + 1];
    t[base]     = (_Float16)(t1 * cs - t2 * sn);
    t[base + 1] = (_Float16)(t1 * sn + t2 * cs);
}

// ---------------------------------------------------------------------------
// Flash-style causal attention. One wave: 16 query rows x full D=64 output.
// q,k: f16 [b][h][s][64] (RoPE'd); vt: f16 [b][h][64][s]; attn: f16 [m][1024]
// block = 128 (4 waves), grid = (S/64, B*H)
// ---------------------------------------------------------------------------
__global__ __launch_bounds__(128) void attn_kernel(
    const _Float16* __restrict__ q, const _Float16* __restrict__ k,
    const _Float16* __restrict__ vt, _Float16* __restrict__ attn)
{
    __shared__ _Float16 plds[4 * 16 * 48];  // per-wave 16x32 P tile, row stride 48
    const int lane = threadIdx.x & 31;
    const int wave = threadIdx.x >> 5;
    const int l16  = lane & 15;
    const int hf   = (lane >> 4) & 1;
    const int bh   = blockIdx.y;                       // b*16 + h
    const int s0   = (blockIdx.x * 4 + wave) * 16;

    const _Float16* qb = q  + (size_t)bh * 2048 * 64;
    const _Float16* kb = k  + (size_t)bh * 2048 * 64;
    const _Float16* vb = vt + (size_t)bh * 64 * 2048;
    _Float16* pw = plds + wave * (16 * 48);

    AF aq0, aq1;
    {
        const _Float16* qrow = qb + (size_t)(s0 + l16) * 64;
        aq0.h[0] = ld8g(qrow + 8 * hf);       aq0.h[1] = ld8g(qrow + 16 + 8 * hf);
        aq1.h[0] = ld8g(qrow + 32 + 8 * hf);  aq1.h[1] = ld8g(qrow + 48 + 8 * hf);
    }

    v8f o[4] = {};
    float mrow[8], lrow[8];
    #pragma unroll
    for (int r = 0; r < 8; ++r) { mrow[r] = -1e30f; lrow[r] = 0.0f; }

    for (int t0 = 0; t0 < s0 + 16; t0 += 32) {
        // prefetch next K/V chunk (global_prefetch_b8; no counter cost)
        if (t0 + 32 < s0 + 16) {
            __builtin_prefetch(kb + (size_t)(t0 + 32 + l16) * 64, 0, 1);
            __builtin_prefetch(kb + (size_t)(t0 + 48 + l16) * 64, 0, 1);
            __builtin_prefetch(vb + (size_t)(lane) * 2048 + t0 + 32, 0, 1);
            __builtin_prefetch(vb + (size_t)(32 + lane) * 2048 + t0 + 32, 0, 1);
        }
        v8f sc0, sc1;
        #pragma unroll
        for (int j = 0; j < 2; ++j) {
            const int tc = t0 + j * 16;
            const _Float16* krow = kb + (size_t)(tc + l16) * 64;
            AF b0, b1;
            b0.h[0] = ld8g(krow + 8 * hf);       b0.h[1] = ld8g(krow + 16 + 8 * hf);
            b1.h[0] = ld8g(krow + 32 + 8 * hf);  b1.h[1] = ld8g(krow + 48 + 8 * hf);
            v8f s = {};
            s = __builtin_amdgcn_wmma_f32_16x16x32_f16(false, aq0.v, false, b0.v, (short)0, s, false, false);
            s = __builtin_amdgcn_wmma_f32_16x16x32_f16(false, aq1.v, false, b1.v, (short)0, s, false, false);
            const int tcol = tc + l16;
            #pragma unroll
            for (int r = 0; r < 8; ++r) {
                int mr = s0 + r + 8 * hf;
                float v = s[r] * 0.125f;               // 1/sqrt(64)
                v = (tcol > mr) ? -1e30f : v;          // causal mask
                if (j == 0) sc0[r] = v; else sc1[r] = v;
            }
        }
        // Online softmax: row stats via xor-swizzle reductions over 16 lanes.
        #pragma unroll
        for (int r = 0; r < 8; ++r) {
            float rm = fmaxf(sc0[r], sc1[r]);
            rm = fmaxf(rm, SWZ_XOR1(rm));
            rm = fmaxf(rm, SWZ_XOR2(rm));
            rm = fmaxf(rm, SWZ_XOR4(rm));
            rm = fmaxf(rm, SWZ_XOR8(rm));
            float mn   = fmaxf(mrow[r], rm);
            float corr = __expf(mrow[r] - mn);
            mrow[r] = mn;
            float p0 = __expf(sc0[r] - mn);
            float p1 = __expf(sc1[r] - mn);
            float rs = p0 + p1;
            rs += SWZ_XOR1(rs);
            rs += SWZ_XOR2(rs);
            rs += SWZ_XOR4(rs);
            rs += SWZ_XOR8(rs);
            lrow[r] = lrow[r] * corr + rs;
            o[0][r] *= corr; o[1][r] *= corr; o[2][r] *= corr; o[3][r] *= corr;
            pw[(r + 8 * hf) * 48 + l16]      = (_Float16)p0;
            pw[(r + 8 * hf) * 48 + 16 + l16] = (_Float16)p1;
        }
        __builtin_amdgcn_wave_barrier();   // LDS is in-order within a wave
        AF ap;
        ap.h[0] = *(const h8*)&pw[l16 * 48 + 8 * hf];
        ap.h[1] = *(const h8*)&pw[l16 * 48 + 16 + 8 * hf];
        __builtin_amdgcn_wave_barrier();
        #pragma unroll
        for (int f = 0; f < 4; ++f) {
            const _Float16* vrow = vb + (size_t)(f * 16 + l16) * 2048 + t0;
            AF bvf;
            bvf.h[0] = ld8g(vrow + 8 * hf);
            bvf.h[1] = ld8g(vrow + 16 + 8 * hf);
            o[f] = __builtin_amdgcn_wmma_f32_16x16x32_f16(
                false, ap.v, false, bvf.v, (short)0, o[f], false, false);
        }
    }

    const int b = bh >> 4, h = bh & 15;
    #pragma unroll
    for (int r = 0; r < 8; ++r) {
        float inv = 1.0f / lrow[r];
        size_t row = (size_t)(b * 2048 + s0 + r + 8 * hf) * 1024 + h * 64;
        attn[row +  0 + l16] = (_Float16)(o[0][r] * inv);
        attn[row + 16 + l16] = (_Float16)(o[1][r] * inv);
        attn[row + 32 + l16] = (_Float16)(o[2][r] * inv);
        attn[row + 48 + l16] = (_Float16)(o[3][r] * inv);
    }
}

// ---------------------------------------------------------------------------
extern "C" void kernel_launch(void* const* d_in, const int* in_sizes, int n_in,
                              void* d_out, int out_size, void* d_ws, size_t ws_size,
                              hipStream_t stream) {
    (void)in_sizes; (void)n_in; (void)out_size; (void)ws_size;
    const float* x  = (const float*)d_in[0];
    const float* Wq = (const float*)d_in[1];
    const float* Wk = (const float*)d_in[2];
    const float* Wv = (const float*)d_in[3];
    const float* bv = (const float*)d_in[4];
    const float* Wo = (const float*)d_in[5];
    const float* bo = (const float*)d_in[6];

    char* ws = (char*)d_ws;
    _Float16* xh  = (_Float16*)(ws);                  //  8 MB  x as f16 [4096][1024]
    _Float16* wqh = (_Float16*)(ws +  8388608);       //  2 MB
    _Float16* wkh = (_Float16*)(ws + 10485760);       //  2 MB
    _Float16* wvh = (_Float16*)(ws + 12582912);       //  2 MB
    _Float16* woh = (_Float16*)(ws + 14680064);       //  2 MB
    _Float16* qh  = (_Float16*)(ws + 16777216);       //  8 MB  [b][h][s][d]
    _Float16* kh  = (_Float16*)(ws + 25165824);       //  8 MB  [b][h][s][d]
    _Float16* vth = (_Float16*)(ws + 33554432);       //  8 MB  [b][h][d][s]
    _Float16* ath = (_Float16*)(ws + 41943040);       //  8 MB  attn out f16 [4096][1024]

    cvt_kernel<<<4096, 256, 0, stream>>>(x,  xh,  4194304);
    cvt_kernel<<<1024, 256, 0, stream>>>(Wq, wqh, 1048576);
    cvt_kernel<<<1024, 256, 0, stream>>>(Wk, wkh, 1048576);
    cvt_kernel<<<1024, 256, 0, stream>>>(Wv, wvh, 1048576);
    cvt_kernel<<<1024, 256, 0, stream>>>(Wo, woh, 1048576);

    dim3 gg(16, 16);   // N/64, M/256
    gemm_wmma_kernel<<<gg, 256, 0, stream>>>(xh, wqh, nullptr, (void*)qh, 0);
    gemm_wmma_kernel<<<gg, 256, 0, stream>>>(xh, wkh, nullptr, (void*)kh, 0);
    gemm_wmma_kernel<<<gg, 256, 0, stream>>>(xh, wvh, bv, (void*)vth, 2);

    rope_kernel<<<8192, 256, 0, stream>>>(qh);
    rope_kernel<<<8192, 256, 0, stream>>>(kh);

    attn_kernel<<<dim3(32, 32), 128, 0, stream>>>(qh, kh, vth, ath);

    gemm_wmma_kernel<<<gg, 256, 0, stream>>>(ath, woh, bo, d_out, 3);
}
